// EncoderV2_79697413145009
// MI455X (gfx1250) — compile-verified, matrix operand order checked by reading, not run on hardware
//
#include <hip/hip_runtime.h>
#include <math.h>

namespace {

constexpr int B = 2, L = 2048, D = 512, H = 8, DH = 2048, Mw = 128, HD = 64;
constexpr int QKV_N = 3 * D;   // 1536
constexpr int ROWS  = B * L;   // 4096
constexpr float EPS = 1e-5f;

typedef __attribute__((ext_vector_type(16))) _Float16 v16h;
typedef __attribute__((ext_vector_type(8)))  _Float16 v8h;
typedef __attribute__((ext_vector_type(8)))  float    v8f;
typedef __attribute__((ext_vector_type(4)))  int      v4i_b;

// ---------------------------------------------------------------- async copy
// 16B global -> LDS async copy (ASYNCcnt-tracked) + constant-count wait.
__device__ __forceinline__ void async_copy16(const _Float16* g, _Float16* l) {
#if __has_builtin(__builtin_amdgcn_global_load_async_to_lds_b128)
  void* gp = (void*)g;
  void* lp = (void*)l;
  __builtin_amdgcn_global_load_async_to_lds_b128(
      (__attribute__((address_space(1))) v4i_b*)gp,
      (__attribute__((address_space(3))) v4i_b*)lp, 0, 0);
#else
  unsigned loff = (unsigned)(unsigned long long)(__attribute__((address_space(3))) void*)(void*)l;
  asm volatile("global_load_async_to_lds_b128 %0, %1, off"
               :: "v"(loff), "v"((unsigned long long)(size_t)g) : "memory");
#endif
}

template <int N>
__device__ __forceinline__ void wait_async() {
#if __has_builtin(__builtin_amdgcn_s_wait_asynccnt)
  __builtin_amdgcn_s_wait_asynccnt(N);
#else
  asm volatile("s_wait_asynccnt %0" :: "i"(N) : "memory");
#endif
}

// ---------------------------------------------------------------- fragments
// f16 fragment straight from global (16B-aligned rows), CDNA5 16-bit layout:
// lane l<16: row l, K {0..7, 16..23}; lane l+16: row l, K {8..15, 24..31}.
__device__ __forceinline__ v16h load_frag_h(const _Float16* __restrict__ base,
                                            int ld, int r0, int k0) {
  const int lane = threadIdx.x & 31;
  const _Float16* p = base + (size_t)(r0 + (lane & 15)) * ld + k0 + ((lane >> 4) << 3);
  v8h lo = *(const v8h*)p;
  v8h hi = *(const v8h*)(p + 16);
  return __builtin_shufflevector(lo, hi, 0, 1, 2, 3, 4, 5, 6, 7,
                                 8, 9, 10, 11, 12, 13, 14, 15);
}

// Same fragment out of an LDS tile with 40-half (80B) padded row stride.
__device__ __forceinline__ v16h lds_frag(const _Float16* s, int row0, int lane) {
  const _Float16* p = s + (row0 + (lane & 15)) * 40 + ((lane >> 4) << 3);
  v8h lo = *(const v8h*)p;
  v8h hi = *(const v8h*)(p + 16);
  return __builtin_shufflevector(lo, hi, 0, 1, 2, 3, 4, 5, 6, 7,
                                 8, 9, 10, 11, 12, 13, 14, 15);
}

// ---------------------------------------------------------------- elementwise
__global__ void transpose_cast_kernel(const float* __restrict__ W,
                                      _Float16* __restrict__ Wt, int K, int N) {
  size_t i = (size_t)blockIdx.x * blockDim.x + threadIdx.x;
  if (i >= (size_t)K * N) return;
  int kk = (int)(i / N), nn = (int)(i % N);
  Wt[(size_t)nn * K + kk] = (_Float16)W[i];
}

__global__ void ln_kernel(const float* __restrict__ x, const float* __restrict__ w,
                          const float* __restrict__ bias, _Float16* __restrict__ out) {
  __shared__ float red[256];
  const int tid = threadIdx.x;
  const float* px = x + (size_t)blockIdx.x * D;
  float v0 = px[tid], v1 = px[tid + 256];
  red[tid] = v0 + v1;
  __syncthreads();
  for (int s = 128; s > 0; s >>= 1) { if (tid < s) red[tid] += red[tid + s]; __syncthreads(); }
  const float mu = red[0] * (1.0f / D);
  __syncthreads();
  const float d0 = v0 - mu, d1 = v1 - mu;
  red[tid] = d0 * d0 + d1 * d1;
  __syncthreads();
  for (int s = 128; s > 0; s >>= 1) { if (tid < s) red[tid] += red[tid + s]; __syncthreads(); }
  const float rs = rsqrtf(red[0] * (1.0f / D) + EPS);
  _Float16* po = out + (size_t)blockIdx.x * D;
  po[tid]       = (_Float16)(d0 * rs * w[tid] + bias[tid]);
  po[tid + 256] = (_Float16)(d1 * rs * w[tid + 256] + bias[tid + 256]);
}

__global__ void split_qkv_kernel(const float* __restrict__ qkv, _Float16* __restrict__ q,
                                 _Float16* __restrict__ k, _Float16* __restrict__ vT) {
  size_t idx = (size_t)blockIdx.x * blockDim.x + threadIdx.x;
  if (idx >= (size_t)ROWS * QKV_N) return;
  int row = (int)(idx / QKV_N), c = (int)(idx % QKV_N);
  int b = row / L, l = row % L;
  int which = c / D, hc = c % D;
  int h = hc / HD, hd = hc % HD;
  _Float16 val = (_Float16)qkv[idx];
  size_t bh = (size_t)b * H + h;
  if (which == 0)      q[(bh * L + l) * HD + hd] = val;
  else if (which == 1) k[(bh * L + l) * HD + hd] = val;
  else                 vT[bh * HD * L + (size_t)hd * L + l] = val;
}

__global__ void geglu_kernel(const float* __restrict__ ab, _Float16* __restrict__ f) {
  size_t idx = (size_t)blockIdx.x * blockDim.x + threadIdx.x;
  if (idx >= (size_t)ROWS * DH) return;
  int row = (int)(idx / DH), c = (int)(idx % DH);
  float a = ab[(size_t)row * (2 * DH) + c];
  float g = ab[(size_t)row * (2 * DH) + DH + c];
  float ge = 0.5f * g * (1.0f + erff(g * 0.70710678118654752f));
  f[idx] = (_Float16)(ge * a);
}

__global__ void residual_kernel(const float* __restrict__ x, const float* __restrict__ o,
                                const float* __restrict__ gamma, float* __restrict__ out) {
  size_t idx = (size_t)blockIdx.x * blockDim.x + threadIdx.x;
  if (idx >= (size_t)ROWS * D) return;
  int c = (int)(idx % D);
  out[idx] = x[idx] + o[idx] * gamma[c];
}

// ---------------------------------------------------------------- staged GEMM
// C[M,N] = A[M,K] @ Bt[N,K]^T (+bias). 256 threads / 8 waves per 128x128 tile.
// Double-buffered: async-DMA prefetch of K-step i+1 overlaps WMMA of step i.
// Steady-state wait is asynccnt<=4 (the 4 in-flight prefetch ops; async loads
// complete in order per wave, so the current tile is guaranteed resident).
__global__ __launch_bounds__(256) void gemm_staged(
    const _Float16* __restrict__ A, int lda,
    const _Float16* __restrict__ Bt, int ldb,
    const float* __restrict__ bias,
    float* __restrict__ C, int ldc, int K) {
  __shared__ _Float16 sA[2][128 * 40];   // 80B row stride (conflict-free b128)
  __shared__ _Float16 sB[2][128 * 40];
  const int tid = threadIdx.x;
  const int wave = tid >> 5, lane = tid & 31;
  const int m0 = blockIdx.y << 7, n0 = blockIdx.x << 7;
  v8f acc[8] = {};

  auto stage = [&](int buf, int k0) {
#pragma unroll
    for (int s = 0; s < 2; ++s) {
      const int ch = tid + (s << 8);        // 512 chunks of 16B per tile
      const int r = ch >> 2, c = (ch & 3) << 3;
      async_copy16(A + (size_t)(m0 + r) * lda + k0 + c, &sA[buf][r * 40 + c]);
      async_copy16(Bt + (size_t)(n0 + r) * ldb + k0 + c, &sB[buf][r * 40 + c]);
    }
  };
  auto compute = [&](int buf) {
    v16h a = lds_frag(sA[buf], wave << 4, lane);
#pragma unroll
    for (int t = 0; t < 8; ++t) {
      v16h b = lds_frag(sB[buf], t << 4, lane);
      acc[t] = __builtin_amdgcn_wmma_f32_16x16x32_f16(false, a, false, b,
                                                      (short)0, acc[t], false, false);
    }
  };

  const int nk = K >> 5;
  stage(0, 0);
  int buf = 0;
  for (int i = 0; i < nk - 1; ++i) {
    stage(buf ^ 1, (i + 1) << 5);   // prefetch next tile into other buffer
    wait_async<4>();                // current tile resident; prefetch in flight
    __syncthreads();
    compute(buf);
    __syncthreads();                // all waves done before buffer is re-staged
    buf ^= 1;
  }
  wait_async<0>();
  __syncthreads();
  compute(buf);

  const int nl = lane & 15;
  const int mb = m0 + (wave << 4) + ((lane >> 4) << 3);
#pragma unroll
  for (int t = 0; t < 8; ++t) {
    const int n = n0 + (t << 4) + nl;
    const float bv = bias ? bias[n] : 0.0f;
#pragma unroll
    for (int j = 0; j < 8; ++j)
      C[(size_t)(mb + j) * ldc + n] = acc[t][j] + bv;
  }
}

// ---------------------------------------------------------------- attention
__global__ void scores_kernel(const _Float16* __restrict__ q, const _Float16* __restrict__ k,
                              const float* __restrict__ rel, float* __restrict__ S) {
  const int z = blockIdx.z;                 // b*H + h
  const int h = z & (H - 1);
  const _Float16* A  = q + (size_t)z * L * HD;
  const _Float16* Bt = k + (size_t)z * L * HD;
  float* C = S + (size_t)z * L * L;
  const int lane = threadIdx.x & 31;
  const int m0 = blockIdx.y << 4;
  const int n0 = blockIdx.x << 6;
  v8f acc[4] = {};
#pragma unroll
  for (int k0 = 0; k0 < HD; k0 += 32) {
    v16h a = load_frag_h(A, HD, m0, k0);
#pragma unroll
    for (int t = 0; t < 4; ++t) {
      v16h b = load_frag_h(Bt, HD, n0 + (t << 4), k0);
      acc[t] = __builtin_amdgcn_wmma_f32_16x16x32_f16(false, a, false, b,
                                                      (short)0, acc[t], false, false);
    }
  }
  const int nl = lane & 15;
  const int mb = m0 + ((lane >> 4) << 3);
  const float* relh = rel + (size_t)h * (2 * Mw - 1);
#pragma unroll
  for (int t = 0; t < 4; ++t) {
    const int n = n0 + (t << 4) + nl;
#pragma unroll
    for (int j = 0; j < 8; ++j) {
      const int i = mb + j;
      int dd = i - n;
      dd = dd > (Mw - 1) ? (Mw - 1) : (dd < -(Mw - 1) ? -(Mw - 1) : dd);
      C[(size_t)i * L + n] = acc[t][j] * 0.125f + relh[dd + (Mw - 1)];
    }
  }
}

__global__ void softmax_kernel(float* __restrict__ attn, _Float16* __restrict__ attn_h) {
  __shared__ float red[256];
  const int tid = threadIdx.x;
  float* p = attn + (size_t)blockIdx.x * L;
  _Float16* ph = attn_h + (size_t)blockIdx.x * L;
  float m = -3.4e38f;
  for (int i = tid; i < L; i += 256) m = fmaxf(m, p[i]);
  red[tid] = m;
  __syncthreads();
  for (int s = 128; s > 0; s >>= 1) { if (tid < s) red[tid] = fmaxf(red[tid], red[tid + s]); __syncthreads(); }
  m = red[0];
  __syncthreads();
  float sum = 0.0f;
  for (int i = tid; i < L; i += 256) { float e = expf(p[i] - m); p[i] = e; sum += e; }
  red[tid] = sum;
  __syncthreads();
  for (int s = 128; s > 0; s >>= 1) { if (tid < s) red[tid] += red[tid + s]; __syncthreads(); }
  const float inv = 1.0f / red[0];
  for (int i = tid; i < L; i += 256) {
    float v = p[i] * inv;
    p[i] = v;
    ph[i] = (_Float16)v;
  }
}

// ctx[b, l, h*64 + n] = sum_k attn[b,h,l,k] * v[b,k,h,n]; ctx stored f16.
__global__ void attnv_kernel(const _Float16* __restrict__ attn_h,
                             const _Float16* __restrict__ vT,
                             _Float16* __restrict__ ctx) {
  const int z = blockIdx.z;
  const int b = z >> 3, h = z & 7;
  const _Float16* A  = attn_h + (size_t)z * L * L;
  const _Float16* Bt = vT + (size_t)z * HD * L;
  _Float16* C = ctx + (size_t)b * L * D + h * HD;
  const int lane = threadIdx.x & 31;
  const int m0 = blockIdx.y << 4;
  v8f acc[4] = {};
  for (int k0 = 0; k0 < L; k0 += 32) {
    v16h a = load_frag_h(A, L, m0, k0);
#pragma unroll
    for (int t = 0; t < 4; ++t) {
      v16h bfr = load_frag_h(Bt, L, (t << 4), k0);
      acc[t] = __builtin_amdgcn_wmma_f32_16x16x32_f16(false, a, false, bfr,
                                                      (short)0, acc[t], false, false);
    }
  }
  const int nl = lane & 15;
  const int mb = m0 + ((lane >> 4) << 3);
#pragma unroll
  for (int t = 0; t < 4; ++t) {
    const int n = (t << 4) + nl;
#pragma unroll
    for (int j = 0; j < 8; ++j)
      C[(size_t)(mb + j) * D + n] = (_Float16)acc[t][j];
  }
}

} // anonymous namespace

extern "C" void kernel_launch(void* const* d_in, const int* in_sizes, int n_in,
                              void* d_out, int out_size, void* d_ws, size_t ws_size,
                              hipStream_t stream) {
  const float* x      = (const float*)d_in[0];
  const float* ln1_w  = (const float*)d_in[1];
  const float* ln1_b  = (const float*)d_in[2];
  const float* qkv_w  = (const float*)d_in[3];
  const float* qkv_b  = (const float*)d_in[4];
  const float* out_w  = (const float*)d_in[5];
  const float* out_b  = (const float*)d_in[6];
  const float* rel_b  = (const float*)d_in[7];
  const float* gamma1 = (const float*)d_in[8];
  const float* ln2_w  = (const float*)d_in[9];
  const float* ln2_b  = (const float*)d_in[10];
  const float* fin_w  = (const float*)d_in[11];
  const float* fin_b  = (const float*)d_in[12];
  const float* fout_w = (const float*)d_in[13];
  const float* fout_b = (const float*)d_in[14];
  const float* gamma2 = (const float*)d_in[15];

  float* x_out = (float*)d_out;                 // [B,L,D]
  float* attn  = x_out + (size_t)B * L * D;     // [B,H,L,L]

  char* w = (char*)d_ws;
  auto grabf = [&](size_t n) { float* p = (float*)w; w += n * sizeof(float); return p; };
  auto grabh = [&](size_t n) { _Float16* p = (_Float16*)w; w += n * sizeof(_Float16); return p; };
  _Float16* wT_qkv  = grabh((size_t)D * QKV_N);
  _Float16* wT_out  = grabh((size_t)D * D);
  _Float16* wT_fin  = grabh((size_t)D * 2 * DH);
  _Float16* wT_fout = grabh((size_t)DH * D);
  _Float16* h1h   = grabh((size_t)ROWS * D);
  _Float16* h2h   = grabh((size_t)ROWS * D);
  _Float16* qh    = grabh((size_t)B * H * L * HD);
  _Float16* kh    = grabh((size_t)B * H * L * HD);
  _Float16* vTh   = grabh((size_t)B * H * HD * L);
  _Float16* attnh = grabh((size_t)B * H * L * L);
  _Float16* ctxh  = grabh((size_t)ROWS * D);
  _Float16* fbufh = grabh((size_t)ROWS * DH);
  float* qkv   = grabf((size_t)ROWS * QKV_N);
  float* oproj = grabf((size_t)ROWS * D);
  float* x1    = grabf((size_t)ROWS * D);
  float* abuf  = grabf((size_t)ROWS * 2 * DH);
  float* ffn   = grabf((size_t)ROWS * D);

  auto cdiv = [](size_t a, size_t b) { return (unsigned)((a + b - 1) / b); };

  // Weight cast+transpose (Bt layout: [N][K], K contiguous, f16).
  transpose_cast_kernel<<<cdiv((size_t)D * QKV_N, 256), 256, 0, stream>>>(qkv_w, wT_qkv, D, QKV_N);
  transpose_cast_kernel<<<cdiv((size_t)D * D, 256), 256, 0, stream>>>(out_w, wT_out, D, D);
  transpose_cast_kernel<<<cdiv((size_t)D * 2 * DH, 256), 256, 0, stream>>>(fin_w, wT_fin, D, 2 * DH);
  transpose_cast_kernel<<<cdiv((size_t)DH * D, 256), 256, 0, stream>>>(fout_w, wT_fout, DH, D);

  // LN1 -> h1 (f16)
  ln_kernel<<<ROWS, 256, 0, stream>>>(x, ln1_w, ln1_b, h1h);

  // QKV projection (double-buffered async-LDS staged WMMA)
  gemm_staged<<<dim3(QKV_N / 128, ROWS / 128), 256, 0, stream>>>(h1h, D, wT_qkv, D, qkv_b, qkv, QKV_N, D);
  split_qkv_kernel<<<cdiv((size_t)ROWS * QKV_N, 256), 256, 0, stream>>>(qkv, qh, kh, vTh);

  // Attention: scores (+rel bias) -> softmax (fp32 in d_out + f16 copy) -> attn@V
  scores_kernel<<<dim3(L / 64, L / 16, B * H), 32, 0, stream>>>(qh, kh, rel_b, attn);
  softmax_kernel<<<B * H * L, 256, 0, stream>>>(attn, attnh);
  attnv_kernel<<<dim3(1, L / 16, B * H), 32, 0, stream>>>(attnh, vTh, ctxh);

  // Output projection + residual 1
  gemm_staged<<<dim3(D / 128, ROWS / 128), 256, 0, stream>>>(ctxh, D, wT_out, D, out_b, oproj, D, D);
  residual_kernel<<<cdiv((size_t)ROWS * D, 256), 256, 0, stream>>>(x, oproj, gamma1, x1);

  // FFN: LN2 -> in-proj -> GEGLU -> out-proj -> residual 2
  ln_kernel<<<ROWS, 256, 0, stream>>>(x1, ln2_w, ln2_b, h2h);
  gemm_staged<<<dim3(2 * DH / 128, ROWS / 128), 256, 0, stream>>>(h2h, D, wT_fin, D, fin_b, abuf, 2 * DH, D);
  geglu_kernel<<<cdiv((size_t)ROWS * DH, 256), 256, 0, stream>>>(abuf, fbufh);
  gemm_staged<<<dim3(D / 128, ROWS / 128), 256, 0, stream>>>(fbufh, DH, wT_fout, DH, fout_b, ffn, D, DH);
  residual_kernel<<<cdiv((size_t)ROWS * D, 256), 256, 0, stream>>>(x1, ffn, gamma2, x_out);

  (void)in_sizes; (void)n_in; (void)out_size; (void)ws_size;
}
